// GGUFQuantizedConv2d_86586540687543
// MI455X (gfx1250) — matrix operand order repaired and checked by reading, not compiled, not run on hardware
//
#include <hip/hip_runtime.h>

// ---------------- problem constants (match reference) ----------------
constexpr int NBATCH = 8;
constexpr int OC     = 256;
constexpr int IC     = 256;
constexpr int HH     = 128;
constexpr int WW     = 128;
constexpr int HW     = HH * WW;          // 16384
constexpr int KFLT   = IC * 9;           // 2304 = GEMM K

// padded channels-last activation tensor: x_pad[nb][h'][w'][ic], h',w' in [0,130)
constexpr int PH = HH + 2;
constexpr int PW = WW + 2;
constexpr size_t WPACK_USHORTS = (size_t)9 * 8 * 16 * 32 * 16;      // 589824
constexpr size_t WPACK_BYTES   = WPACK_USHORTS * 2;                 // 1,179,648
constexpr size_t XPAD_USHORTS  = (size_t)NBATCH * PH * PW * IC;     // 34,611,200
constexpr size_t XPAD_BYTES    = XPAD_USHORTS * 2;                  // ~69.2 MB

typedef __attribute__((ext_vector_type(16))) __bf16       bf16x16;
typedef __attribute__((ext_vector_type(8)))  float        f32x8;
typedef __attribute__((ext_vector_type(8)))  unsigned int u32x8;

__device__ __forceinline__ unsigned short f2bf(float f) {
  // round-to-nearest-even f32 -> bf16 bits
  unsigned int u = __builtin_bit_cast(unsigned int, f);
  u += 0x7FFFu + ((u >> 16) & 1u);
  return (unsigned short)(u >> 16);
}

// =====================================================================
// Pre-pass 1: dequantize GGUF-Q8 weights into A-fragment layout
// (16-bit A 16x32, contiguous-K order, B uses the same): lane l: M=l&15, K = {kb..kb+15},
// kb=(l>>4)*16.  Pack index: ((khw*8+icc)*16 + ocg)*32 + lane -> 16 bf16.
// =====================================================================
__global__ __launch_bounds__(256) void dequant_pack_w(
    const int*   __restrict__ qw,
    const float* __restrict__ scales,
    unsigned short* __restrict__ wpack)
{
  const int t    = blockIdx.x * 256 + threadIdx.x;   // 0..36863
  const int lane = t & 31;
  const int ocg  = (t >> 5) & 15;
  const int icc  = (t >> 9) & 7;
  const int khw  = t >> 12;                          // 0..8
  const int oc   = ocg * 16 + (lane & 15);
  const int kb   = (lane >> 4) * 16;

  u32x8 pk;
#pragma unroll
  for (int p = 0; p < 8; ++p) {
    unsigned int w = 0;
#pragma unroll
    for (int s = 0; s < 2; ++s) {
      const int e  = 2 * p + s;
      const int kk = kb + e;   // K runs {kb..kb+15}
      const int flat = oc * KFLT + (icc * 32 + kk) * 9 + khw;
      const float v  = (float)(qw[flat] - 64) * scales[flat >> 5];
      w |= ((unsigned int)f2bf(v)) << (16 * s);
    }
    pk[p] = w;
  }
  *(u32x8*)(wpack + (size_t)t * 16) = pk;
}

// =====================================================================
// Pre-pass 2: transpose x NCHW-f32 -> padded NHWC-bf16 (interior only;
// halo is zeroed by hipMemsetAsync). One block per (nb,h) row.
// =====================================================================
__global__ __launch_bounds__(256) void pack_x_nhwc(
    const float* __restrict__ x, unsigned short* __restrict__ xpad)
{
  const int row = blockIdx.x;              // 0..1023
  const int nb  = row >> 7;
  const int h   = row & 127;
  const int px  = threadIdx.x & 127;
  const int ich = (threadIdx.x >> 7) * 16; // 0 / 16

  const float* src = x + (size_t)nb * IC * HW + (size_t)h * WW + px;
  unsigned short* dst = xpad + (((size_t)(nb * PH + h + 1)) * PW + px + 1) * IC;

#pragma unroll
  for (int g = 0; g < 8; ++g) {
    const int ic0 = g * 32 + ich;
    u32x8 pk;
#pragma unroll
    for (int p = 0; p < 8; ++p) {
      const float v0 = src[(size_t)(ic0 + 2 * p) * HW];
      const float v1 = src[(size_t)(ic0 + 2 * p + 1) * HW];
      pk[p] = (unsigned int)f2bf(v0) | ((unsigned int)f2bf(v1) << 16);
    }
    *(u32x8*)(dst + ic0) = pk;
  }
}

// =====================================================================
// Main kernel: implicit GEMM conv, LDS-free / barrier-free.
// Wave tile: 32 oc x 64 px (2 A-strips x 4 B-groups), 8 accumulators.
// Per K-step: 4 A-loads + 8 B-loads (global_load_b128) feed 8 WMMAs
// (1.5 loads/WMMA; each B frag used by 2 WMMAs, each A frag by 4).
// Workgroup = 8 waves = 4 oc-pairs x 2 px-halves; 2 WGs per image row.
// =====================================================================
__global__ __launch_bounds__(256) void conv_q8_wmma_bf16(
    const unsigned short* __restrict__ xpad,
    const unsigned short* __restrict__ wpack,
    const float* __restrict__ bias,
    float* __restrict__ out)
{
  const int tid  = threadIdx.x;
  const int lane = tid & 31;
  const int wave = tid >> 5;                         // 0..7
  const int ocg0 = ((blockIdx.x & 1) * 4 + (wave >> 1)) * 2;  // first 16-oc strip
  const int ph   = wave & 1;                         // px half (64 px each)
  const int row  = blockIdx.x >> 1;                  // 0..1023
  const int nb   = row >> 7;
  const int h    = row & 127;

  const int nL  = lane & 15;           // B column within 16-px group
  const int kbB = (lane >> 4) * 16;    // B K-run: 0..15 / 16..31

  f32x8 acc[2][4];
#pragma unroll
  for (int s = 0; s < 2; ++s)
#pragma unroll
    for (int c = 0; c < 4; ++c)
      acc[s][c] = f32x8{0.f,0.f,0.f,0.f,0.f,0.f,0.f,0.f};

  // per-lane bases
  const unsigned short* bbase =
      xpad + (((size_t)(nb * PH + h)) * PW + ph * 64 + nL) * IC + kbB;
  const unsigned short* abase = wpack + ((size_t)ocg0 * 32 + lane) * 16;

  for (int khw = 0; khw < 9; ++khw) {
    const int kh = khw / 3;
    const int kw = khw - 3 * kh;
    const unsigned short* bk = bbase + ((size_t)kh * PW + kw) * IC;
    const unsigned short* ak = abase + (size_t)khw * 8 * 8192;  // 8 icc * 16*32*16

    // warm the next tap's row while this one computes
    if (khw < 8)
      __builtin_prefetch(bk + (size_t)PW * IC, 0, 1);

#pragma unroll
    for (int icc = 0; icc < 8; ++icc) {
      const unsigned short* bi = bk + icc * 32;
      const unsigned short* ai = ak + (size_t)icc * 8192;

      // A fragments: 2 oc-strips (adjacent ocg: +512 ushorts = 1 KB)
      const bf16x16 aF0 = __builtin_bit_cast(bf16x16, *(const u32x8*)(ai));
      const bf16x16 aF1 = __builtin_bit_cast(bf16x16, *(const u32x8*)(ai + 512));

      // B fragments: 4 px-groups; each feeds both A strips
#pragma unroll
      for (int c = 0; c < 4; ++c) {
        const bf16x16 bF = __builtin_bit_cast(
            bf16x16, *(const u32x8*)(bi + (size_t)c * 16 * IC));
        acc[0][c] = __builtin_amdgcn_wmma_f32_16x16x32_bf16(
                        false, aF0, false, bF, (short)0, acc[0][c], false, false);
        acc[1][c] = __builtin_amdgcn_wmma_f32_16x16x32_bf16(
                        false, aF1, false, bF, (short)0, acc[1][c], false, false);
      }
    }
  }

  // ---- epilogue: C layout (VGPR i -> M=(lane>>4)*8+i, N=lane&15) ----
  // Non-temporal stores: out is write-once, keep x_pad/weights in L2.
  const int mh = (lane >> 4) * 8;
#pragma unroll
  for (int s = 0; s < 2; ++s) {
#pragma unroll
    for (int i = 0; i < 8; ++i) {
      const int oc  = (ocg0 + s) * 16 + mh + i;
      const float b = bias[oc];
      float* orow = out + ((size_t)nb * OC + oc) * HW + (size_t)h * WW
                        + ph * 64 + nL;
#pragma unroll
      for (int c = 0; c < 4; ++c)
        __builtin_nontemporal_store(acc[s][c][i] + b, &orow[c * 16]);
    }
  }
}

extern "C" void kernel_launch(void* const* d_in, const int* in_sizes, int n_in,
                              void* d_out, int out_size, void* d_ws, size_t ws_size,
                              hipStream_t stream) {
  const float* x      = (const float*)d_in[0];
  const int*   qw     = (const int*)  d_in[1];
  const float* scales = (const float*)d_in[2];
  const float* bias   = (const float*)d_in[3];
  float*       out    = (float*)d_out;

  unsigned short* wpack = (unsigned short*)d_ws;
  unsigned short* xpad  = (unsigned short*)((char*)d_ws + WPACK_BYTES);

  // zero the halo (and everything else; interior rewritten below)
  hipMemsetAsync(xpad, 0, XPAD_BYTES, stream);

  // 1) weights -> A-fragment bf16 layout (1.18 MB, L2-resident)
  dequant_pack_w<<<144, 256, 0, stream>>>(qw, scales, wpack);

  // 2) x -> padded NHWC bf16 (one-time conversion instead of 36x in hot loop)
  pack_x_nhwc<<<NBATCH * HH, 256, 0, stream>>>(x, xpad);

  // 3) barrier-free implicit-GEMM conv: 2 WGs per image row x 1024 rows
  conv_q8_wmma_bf16<<<2 * NBATCH * HH, 256, 0, stream>>>(xpad, wpack, bias, out);
}
